// ClusterMemory_78984448573994
// MI455X (gfx1250) — matrix operand order repaired and checked by reading, not compiled, not run on hardware
//
#include <hip/hip_runtime.h>
#include <hip/hip_bf16.h>
#include <math.h>

typedef __attribute__((ext_vector_type(16))) _Float16 v16h;
typedef __attribute__((ext_vector_type(8)))  float    v8f;
typedef __attribute__((ext_vector_type(4)))  float    v4f;

// builtin's expected pointee type (from hipcc diagnostic): int __vector_size__(16)
typedef int v4i_vs __attribute__((__vector_size__(16)));
typedef __attribute__((address_space(1))) v4i_vs* as1_v4i;
typedef __attribute__((address_space(3))) v4i_vs* as3_v4i;

#define NSAMP   100000
#define NF      256
#define NTILE   64
#define NCHUNK  ((NSAMP + NTILE - 1) / NTILE)   // 1563
#define INVT    20.0f                            // 1 / 0.05

#if defined(__AMDGCN__) && __has_builtin(__builtin_amdgcn_global_load_async_to_lds_b128)
#define USE_ASYNC_LDS 1
#else
#define USE_ASYNC_LDS 0
#endif

// ---------------- helpers ----------------
__device__ __forceinline__ void atomicMaxF(float* addr, float val) {
    int* a = (int*)addr;
    int cur = __float_as_int(*addr);
    while (__int_as_float(cur) < val) {
        int prev = atomicCAS(a, cur, __float_as_int(val));
        if (prev == cur) break;
        cur = prev;
    }
}

// Stage 4 KB-strided 16 B pieces (64 B per thread, 16 KB per 256-thread block)
__device__ __forceinline__ void stage_issue(const char* g, char* l, int tid) {
#if USE_ASYNC_LDS
#pragma unroll
    for (int i = 0; i < 4; ++i)
        __builtin_amdgcn_global_load_async_to_lds_b128(
            (as1_v4i)(void*)(g + tid * 16 + i * 4096),
            (as3_v4i)(void*)(l + tid * 16 + i * 4096),
            0, 0);
#else
#pragma unroll
    for (int i = 0; i < 4; ++i)
        *(v4f*)(l + tid * 16 + i * 4096) = *(const v4f*)(g + tid * 16 + i * 4096);
#endif
}

__device__ __forceinline__ void stage_wait(int pending) {
#if USE_ASYNC_LDS
    if (pending) asm volatile("s_wait_asynccnt 0x4" ::: "memory");
    else         asm volatile("s_wait_asynccnt 0x0" ::: "memory");
#else
    (void)pending;
#endif
}

// ---------------- 1) normalize + pack A (f16, WMMA lane layout) ----------------
// apack[((bank*8 + kb)*2 + g)*4096 + m*16 + s]
//   g = lane-half; slot s -> k_local = 16*g + s  (contiguous k order, matches B)
__global__ __launch_bounds__(256)
void prep_kernel(const float* __restrict__ i0, const float* __restrict__ i1,
                 const float* __restrict__ i2, float* __restrict__ xnorm,
                 _Float16* __restrict__ apack) {
    int bank = blockIdx.x >> 8;
    int b    = blockIdx.x & 255;
    int t    = threadIdx.x;
    const float* src = bank == 0 ? i0 : (bank == 1 ? i1 : i2);
    float v = src[(size_t)b * NF + t];
    __shared__ float red[256];
    red[t] = v * v; __syncthreads();
    for (int s = 128; s; s >>= 1) { if (t < s) red[t] += red[t + s]; __syncthreads(); }
    float scale = 1.0f / fmaxf(sqrtf(red[0]), 1e-12f);
    float xn = v * scale;
    xnorm[((size_t)bank * 256 + b) * NF + t] = xn;
    int kb = t >> 5, kl = t & 31, blk = kl >> 3;
    int g  = blk >> 1;
    int s2 = (kl & 7) + 8 * (blk & 1);
    apack[(((size_t)bank * 8 + kb) * 2 + g) * 4096 + b * 16 + s2] = (_Float16)xn;
}

// ---------------- 2) JS term on softmaxed normalized x1/x2 ----------------
__global__ __launch_bounds__(256)
void js_kernel(const float* __restrict__ xnorm, float* __restrict__ jsp) {
    int b = blockIdx.x, t = threadIdx.x;
    __shared__ float red[256];
    float a1 = xnorm[(size_t)(256 + b) * NF + t];
    float a2 = xnorm[(size_t)(512 + b) * NF + t];

    red[t] = a1; __syncthreads();
    for (int s = 128; s; s >>= 1) { if (t < s) red[t] = fmaxf(red[t], red[t + s]); __syncthreads(); }
    float m1 = red[0]; __syncthreads();
    float e1 = __expf(a1 - m1);
    red[t] = e1; __syncthreads();
    for (int s = 128; s; s >>= 1) { if (t < s) red[t] += red[t + s]; __syncthreads(); }
    float p1 = e1 / red[0]; __syncthreads();

    red[t] = a2; __syncthreads();
    for (int s = 128; s; s >>= 1) { if (t < s) red[t] = fmaxf(red[t], red[t + s]); __syncthreads(); }
    float m2 = red[0]; __syncthreads();
    float e2 = __expf(a2 - m2);
    red[t] = e2; __syncthreads();
    for (int s = 128; s; s >>= 1) { if (t < s) red[t] += red[t + s]; __syncthreads(); }
    float p2 = e2 / red[0]; __syncthreads();

    float mm   = 0.5f * (p1 + p2);
    float lmm  = __logf(mm);
    float term = p1 * (__logf(p1) - lmm) + p2 * (__logf(p2) - lmm);
    red[t] = term; __syncthreads();
    for (int s = 128; s; s >>= 1) { if (t < s) red[t] += red[t + s]; __syncthreads(); }
    if (t == 0) jsp[b] = 0.5f * red[0];
}

// ---------------- 3) exact f32 target-column logits ----------------
__global__ __launch_bounds__(256)
void tlogit_kernel(const float* __restrict__ xnorm, const int* __restrict__ targets,
                   const float* __restrict__ f0, const float* __restrict__ f1,
                   const float* __restrict__ f2, float* __restrict__ tlogit) {
    int r    = blockIdx.x * 8 + (threadIdx.x >> 5);
    int lane = threadIdx.x & 31;
    int bank = r >> 8, b = r & 255;
    const float* feat = bank == 0 ? f0 : (bank == 1 ? f1 : f2);
    int tgt = targets[b];
    const float* xr = xnorm + (size_t)r * NF;
    const float* fr = feat  + (size_t)tgt * NF;
    float s = 0.f;
    for (int k = lane; k < NF; k += 32) s += xr[k] * fr[k];
    s += __shfl_xor(s, 1, 32);  s += __shfl_xor(s, 2, 32);
    s += __shfl_xor(s, 4, 32);  s += __shfl_xor(s, 8, 32);
    s += __shfl_xor(s, 16, 32);
    if (lane == 0) tlogit[r] = s * INVT;
}

// ---------------- 4) big WMMA GEMM + per-chunk row max/sumexp ----------------
__global__ __launch_bounds__(256)
void gemm_lse_kernel(const float* __restrict__ f0, const float* __restrict__ f1,
                     const float* __restrict__ f2, const _Float16* __restrict__ apack,
                     float* __restrict__ pmax, float* __restrict__ psum) {
    const int bank  = blockIdx.y;
    const int chunk = blockIdx.x;
    const float* feat = bank == 0 ? f0 : (bank == 1 ? f1 : f2);
    const int tid   = threadIdx.x;
    const int wave  = tid >> 5;
    const int lane  = tid & 31;
    const int grp   = lane >> 4;          // lane half
    const int l15   = lane & 15;
    const int nsub  = wave >> 1;          // 4 N-subtiles of 16
    const int mhalf = wave & 1;           // rows [0,128) or [128,256)
    const int n = chunk * NTILE + nsub * 16 + l15;
    const bool nvalid = n < NSAMP;

    __shared__ __align__(32) _Float16 sA[2][2 * 256 * 16];  // double-buffered 16 KB A tile
    __shared__ float lmax[256];
    __shared__ float lsum[256];

    v8f zero = {};
    v8f acc[8];
#pragma unroll
    for (int i = 0; i < 8; ++i) acc[i] = zero;

    const char* asrc = (const char*)(apack + (size_t)bank * (8 * 2 * 256 * 16));
    const float* fbase = feat + (size_t)n * NF + grp * 16;   // lane's B stream base

    // prologue: stage A K-block 0 (async -> LDS) and prefetch B K-block 0 (regs)
    stage_issue(asrc, (char*)&sA[0][0], tid);
    v4f q[4] = {v4f{}, v4f{}, v4f{}, v4f{}};
    if (nvalid) {
        const v4f* fp = (const v4f*)fbase;
#pragma unroll
        for (int i = 0; i < 4; ++i) q[i] = __builtin_nontemporal_load(fp + i);
    }

    for (int kb = 0; kb < 8; ++kb) {
        const int buf = kb & 1;
        if (kb < 7) {  // overlap next A staging with this block's WMMAs
            stage_issue(asrc + (kb + 1) * 16384, (char*)&sA[buf ^ 1][0], tid);
            stage_wait(1);
        } else {
            stage_wait(0);
        }
        __syncthreads();

        // convert prefetched B(kb) f32 -> f16 fragment
        v16h bm;
#pragma unroll
        for (int i = 0; i < 4; ++i) {
            bm[4 * i + 0] = (_Float16)q[i].x; bm[4 * i + 1] = (_Float16)q[i].y;
            bm[4 * i + 2] = (_Float16)q[i].z; bm[4 * i + 3] = (_Float16)q[i].w;
        }

        // prefetch B(kb+1): NT global loads issued before the WMMA block
        v4f qn[4] = {v4f{}, v4f{}, v4f{}, v4f{}};
        if (kb < 7 && nvalid) {
            const v4f* fp = (const v4f*)(fbase + (kb + 1) * 32);
#pragma unroll
            for (int i = 0; i < 4; ++i) qn[i] = __builtin_nontemporal_load(fp + i);
        }

        // preload all 8 A fragments (one dscnt wait), then 8 back-to-back WMMAs
        const v16h* pA = (const v16h*)&sA[buf][0];
        v16h am[8];
#pragma unroll
        for (int mt = 0; mt < 8; ++mt)
            am[mt] = pA[grp * 256 + mhalf * 128 + mt * 16 + l15];
#pragma unroll
        for (int mt = 0; mt < 8; ++mt)
            acc[mt] = __builtin_amdgcn_wmma_f32_16x16x32_f16(
                false, am[mt], false, bm, (short)0, acc[mt], false, false);

#pragma unroll
        for (int i = 0; i < 4; ++i) q[i] = qn[i];
        __syncthreads();   // protect buf from being overwritten by kb+2's staging
    }

    // D layout: lane l holds column n=(l&15), rows m = 16*mt + vidx + 8*(l>>4) (+ mhalf*128)
    lmax[tid] = -INFINITY;
    __syncthreads();
#pragma unroll
    for (int mt = 0; mt < 8; ++mt) {
#pragma unroll
        for (int j = 0; j < 8; ++j) {
            float v = nvalid ? acc[mt][j] * INVT : -INFINITY;
            v = fmaxf(v, __shfl_xor(v, 1, 16));
            v = fmaxf(v, __shfl_xor(v, 2, 16));
            v = fmaxf(v, __shfl_xor(v, 4, 16));
            v = fmaxf(v, __shfl_xor(v, 8, 16));
            if (l15 == 0) {
                int m = mhalf * 128 + mt * 16 + j + 8 * grp;
                atomicMaxF(&lmax[m], v);
            }
        }
    }
    __syncthreads();
    lsum[tid] = 0.0f;
    __syncthreads();
#pragma unroll
    for (int mt = 0; mt < 8; ++mt) {
        int mbase = mhalf * 128 + mt * 16 + 8 * grp;
#pragma unroll
        for (int j = 0; j < 8; ++j) {
            int m = mbase + j;
            float e = nvalid ? __expf(acc[mt][j] * INVT - lmax[m]) : 0.0f;
            e += __shfl_xor(e, 1, 16);
            e += __shfl_xor(e, 2, 16);
            e += __shfl_xor(e, 4, 16);
            e += __shfl_xor(e, 8, 16);
            if (l15 == 0) atomicAdd(&lsum[m], e);   // ds_add_f32
        }
    }
    __syncthreads();
    size_t po = ((size_t)bank * NCHUNK + chunk) * 256 + tid;
    pmax[po] = lmax[tid];
    psum[po] = lsum[tid];
}

// ---------------- 5) combine chunk partials -> row LSE ----------------
__global__ __launch_bounds__(256)
void lse_reduce_kernel(const float* __restrict__ pmax, const float* __restrict__ psum,
                       float* __restrict__ lse) {
    int r    = blockIdx.x * 8 + (threadIdx.x >> 5);
    int lane = threadIdx.x & 31;
    int bank = r >> 8, b = r & 255;
    const float* pm = pmax + (size_t)bank * NCHUNK * 256 + b;
    const float* ps = psum + (size_t)bank * NCHUNK * 256 + b;
    float M = -INFINITY;
    for (int c = lane; c < NCHUNK; c += 32) M = fmaxf(M, pm[(size_t)c * 256]);
    for (int o = 16; o; o >>= 1) M = fmaxf(M, __shfl_xor(M, o, 32));
    float S = 0.f;
    for (int c = lane; c < NCHUNK; c += 32)
        S += ps[(size_t)c * 256] * __expf(pm[(size_t)c * 256] - M);
    for (int o = 16; o; o >>= 1) S += __shfl_xor(S, o, 32);
    if (lane == 0) lse[r] = M + __logf(S);
}

// ---------------- 6) final scalar ----------------
__global__ __launch_bounds__(256)
void final_kernel(const float* __restrict__ lse, const float* __restrict__ tlogit,
                  const float* __restrict__ jsp, float* __restrict__ out) {
    __shared__ float red[256];
    int t = threadIdx.x;
    float ce = 0.f;
    for (int r = t; r < 768; r += 256) ce += lse[r] - tlogit[r];
    red[t] = ce * (1.0f / 256.0f) + jsp[t];
    __syncthreads();
    for (int s = 128; s; s >>= 1) { if (t < s) red[t] += red[t + s]; __syncthreads(); }
    if (t == 0) out[0] = red[0];
}

// ---------------- launch ----------------
extern "C" void kernel_launch(void* const* d_in, const int* in_sizes, int n_in,
                              void* d_out, int out_size, void* d_ws, size_t ws_size,
                              hipStream_t stream) {
    const float* inp0 = (const float*)d_in[0];
    const float* inp1 = (const float*)d_in[1];
    const float* inp2 = (const float*)d_in[2];
    const int*   tgt  = (const int*)  d_in[3];
    const float* f0   = (const float*)d_in[4];
    const float* f1   = (const float*)d_in[5];
    const float* f2   = (const float*)d_in[6];

    char* ws = (char*)d_ws;
    float*    xnorm = (float*)   (ws + 0);          // 786432 B
    _Float16* apack = (_Float16*)(ws + 786432);     // 393216 B
    float*    pmax  = (float*)   (ws + 1179648);    // 4801536 B
    float*    psum  = (float*)   (ws + 5981184);    // 4801536 B
    float*    tl    = (float*)   (ws + 10782720);   // 3072 B
    float*    lse   = (float*)   (ws + 10785792);   // 3072 B
    float*    jsp   = (float*)   (ws + 10788864);   // 1024 B  (total ~10.3 MB)
    float*    out   = (float*)d_out;

    prep_kernel      <<<768, 256, 0, stream>>>(inp0, inp1, inp2, xnorm, apack);
    js_kernel        <<<256, 256, 0, stream>>>(xnorm, jsp);
    tlogit_kernel    <<<96,  256, 0, stream>>>(xnorm, tgt, f0, f1, f2, tl);
    gemm_lse_kernel  <<<dim3(NCHUNK, 3), 256, 0, stream>>>(f0, f1, f2, apack, pmax, psum);
    lse_reduce_kernel<<<96,  256, 0, stream>>>(pmax, psum, lse);
    final_kernel     <<<1,   256, 0, stream>>>(lse, tl, jsp, out);
}
